// AttentionHead_15805479649975
// MI455X (gfx1250) — matrix operand order repaired and checked by reading, not compiled, not run on hardware
//
#include <hip/hip_runtime.h>
#include <cstdint>
#include <cstddef>

typedef __attribute__((ext_vector_type(16))) _Float16 v16h;
typedef __attribute__((ext_vector_type(2)))  _Float16 h2;
typedef __attribute__((ext_vector_type(8)))  float    v8f;

#define D_MODEL 1024
#define D_K     64
#define BATCH   8
#define SEQ     2048
#define NROWS   (BATCH * SEQ)   // 16384

// --- WMMA fragment K-offset patterns (CDNA5 ISA 7.12.2, 16-bit, wave32) ---
// A (16x32): VGPR v<4 holds K=2v,2v+1 ; v>=4 holds K=16+2(v-4),.. ; +8 for lanes 16-31
__device__ __forceinline__ int kpatA(int v, int hi) {
    return (v < 4 ? 2 * v : 16 + 2 * (v - 4)) + (hi ? 8 : 0);
}
// contiguous alt, unused (B loads use kpatA): VGPR v holds K=2v,2v+1 for lanes 0-15 ; K=16+2v,.. for lanes 16-31
__device__ __forceinline__ int kpatB(int v, int hi) {
    return 2 * v + (hi ? 16 : 0);
}

__device__ __forceinline__ v8f wmma_f16(v16h a, v16h b, v8f c) {
    return __builtin_amdgcn_wmma_f32_16x16x32_f16(false, a, false, b, (short)0, c,
                                                  false, false);
}

// ============================================================================
// Kernel 1: fused QKV projection.  [16384 x 1024] @ [1024 x 64] + bias, x3.
// Weight phase loop fully unrolled so each phase's epilogue is straight-line
// code (no runtime pointer muxing / branches).  Weights staged transposed in
// LDS as f16 (two 64KB K-chunks).  Q,K stored f16 [B,S,64]; V stored f16
// transposed [B,64,S] so attention B-fragments are pair-contiguous.
// ============================================================================
__global__ __launch_bounds__(128)
void qkv_proj_kernel(const float* __restrict__ x,
                     const float* __restrict__ Wq, const float* __restrict__ bq,
                     const float* __restrict__ Wk, const float* __restrict__ bk,
                     const float* __restrict__ Wv, const float* __restrict__ bv,
                     _Float16* __restrict__ Qh, _Float16* __restrict__ Kh,
                     _Float16* __restrict__ Vt)
{
    __shared__ _Float16 WtLDS[D_K * 512];          // 64 KB, [n][kk] transposed chunk

    const int tid  = threadIdx.x;
    const int wave = tid >> 5;
    const int lane = tid & 31;
    const int hi   = lane >> 4;
    const int lo   = lane & 15;
    const int m0   = blockIdx.x * 64 + wave * 16;  // 16 output rows per wave

    const float* Ws[3] = {Wq, Wk, Wv};
    const float* Bs[3] = {bq, bk, bv};

    #pragma unroll
    for (int w = 0; w < 3; ++w) {                  // compile-time phase
        v8f acc[4];
        #pragma unroll
        for (int j = 0; j < 4; ++j) acc[j] = (v8f)0.0f;

        #pragma unroll
        for (int kc = 0; kc < 2; ++kc) {           // two 512-deep K chunks
            __syncthreads();
            const float* W = Ws[w];
            // Cooperative transpose stage: float4 over fast-varying n (coalesced).
            for (int it = 0; it < 64; ++it) {
                int idx4 = (tid + it * 128) * 4;   // element index, n-fast
                int n    = idx4 & (D_K - 1);
                int kk   = idx4 >> 6;
                float4 f = *(const float4*)(W + (size_t)(kc * 512 + kk) * D_K + n);
                WtLDS[(n + 0) * 512 + kk] = (_Float16)f.x;
                WtLDS[(n + 1) * 512 + kk] = (_Float16)f.y;
                WtLDS[(n + 2) * 512 + kk] = (_Float16)f.z;
                WtLDS[(n + 3) * 512 + kk] = (_Float16)f.w;
            }
            __syncthreads();

            for (int ks = 0; ks < 16; ++ks) {      // K-steps of 32
                v16h a;
                const int row   = m0 + lo;
                const int dbase = kc * 512 + ks * 32;
                #pragma unroll
                for (int v = 0; v < 8; ++v) {
                    int d = dbase + kpatA(v, hi);
                    float2 f = *(const float2*)(x + (size_t)row * D_MODEL + d);
                    a[2 * v]     = (_Float16)f.x;
                    a[2 * v + 1] = (_Float16)f.y;
                }
                #pragma unroll
                for (int j = 0; j < 4; ++j) {
                    v16h bf;
                    int n = 16 * j + lo;
                    #pragma unroll
                    for (int v = 0; v < 8; ++v) {
                        int kk = ks * 32 + kpatA(v, hi);
                        h2 p = *(const h2*)&WtLDS[n * 512 + kk];
                        bf[2 * v]     = p[0];
                        bf[2 * v + 1] = p[1];
                    }
                    acc[j] = wmma_f16(a, bf, acc[j]);
                }
            }
        }

        // Epilogue (w is compile-time): bias add, f16 convert, store.
        const float* bias = Bs[w];
        #pragma unroll
        for (int j = 0; j < 4; ++j) {
            int   n  = 16 * j + lo;
            float bb = bias[n];
            #pragma unroll
            for (int r = 0; r < 8; ++r) {
                int   row = m0 + r + hi * 8;       // C layout: M = r + 8*hi
                float val = acc[j][r] + bb;
                if (w == 0)      Qh[(size_t)row * D_K + n] = (_Float16)val;
                else if (w == 1) Kh[(size_t)row * D_K + n] = (_Float16)val;
                else {
                    int batch = row >> 11, s = row & (SEQ - 1);
                    Vt[((size_t)batch * D_K + n) * SEQ + s] = (_Float16)val;
                }
            }
        }
    }
}

// ============================================================================
// Kernel 2: fused attention.  Online softmax over ALL keys (reference applies
// causal+padding masks AFTER softmax), masked numerator accumulation, pad(q)
// and 1/Z applied in the epilogue.  One 16-row q tile per wave.
// ============================================================================
__global__ __launch_bounds__(128)
void attn_kernel(const _Float16* __restrict__ Qh, const _Float16* __restrict__ Kh,
                 const _Float16* __restrict__ Vt, const int* __restrict__ pad_all,
                 float* __restrict__ out)
{
    __shared__ _Float16 Plds[4][16 * 32];          // per-wave P-transpose buffer

    const int tid  = threadIdx.x;
    const int wave = tid >> 5;
    const int lane = tid & 31;
    const int hi   = lane >> 4;
    const int lo   = lane & 15;
    const int b    = blockIdx.x >> 5;
    const int q0   = (blockIdx.x & 31) * 64 + wave * 16;

    const _Float16* Q   = Qh + (size_t)b * SEQ * D_K;
    const _Float16* K   = Kh + (size_t)b * SEQ * D_K;
    const _Float16* V   = Vt + (size_t)b * D_K * SEQ;
    const int*      pad = pad_all + b * SEQ;
    _Float16*       P   = &Plds[wave][0];

    // Q A-fragments for d=0..31 and d=32..63 (resident for whole kernel)
    v16h aQ0, aQ1;
    {
        const h2* qp = (const h2*)(Q + (size_t)(q0 + lo) * D_K);
        #pragma unroll
        for (int v = 0; v < 8; ++v) {
            int d  = kpatA(v, hi);
            h2  p0 = qp[d >> 1];
            h2  p1 = qp[(32 + d) >> 1];
            aQ0[2 * v] = p0[0]; aQ0[2 * v + 1] = p0[1];
            aQ1[2 * v] = p1[0]; aQ1[2 * v + 1] = p1[1];
        }
    }

    float mrun[8], Zrun[8];
    v8f   acc[4];
    #pragma unroll
    for (int r = 0; r < 8; ++r) { mrun[r] = -3.0e38f; Zrun[r] = 0.0f; }
    #pragma unroll
    for (int j = 0; j < 4; ++j) acc[j] = (v8f)0.0f;

    for (int kb = 0; kb < SEQ; kb += 32) {
        // Prefetch next key block (K rows + V rows) while we compute this one.
        {
            int kbn = (kb + 32 < SEQ) ? kb + 32 : kb;
            __builtin_prefetch(K + (size_t)(kbn + lane) * D_K, 0, 3);
            __builtin_prefetch(V + (size_t)(2 * lane) * SEQ + kbn, 0, 3);
        }

        // ---- scores: two 16-key tiles, K-dim 64 = two WMMAs each ----
        v8f c0 = (v8f)0.0f, c1 = (v8f)0.0f;
        {
            const h2* kp = (const h2*)(K + (size_t)(kb + lo) * D_K);
            v16h b0, b1;
            #pragma unroll
            for (int v = 0; v < 8; ++v) {
                int d  = kpatA(v, hi);               // B holds K^T: d-major pairs
                h2  p0 = kp[d >> 1];
                h2  p1 = kp[(32 + d) >> 1];
                b0[2 * v] = p0[0]; b0[2 * v + 1] = p0[1];
                b1[2 * v] = p1[0]; b1[2 * v + 1] = p1[1];
            }
            c0 = wmma_f16(aQ0, b0, c0);
            c0 = wmma_f16(aQ1, b1, c0);
        }
        {
            const h2* kp = (const h2*)(K + (size_t)(kb + 16 + lo) * D_K);
            v16h b0, b1;
            #pragma unroll
            for (int v = 0; v < 8; ++v) {
                int d  = kpatA(v, hi);
                h2  p0 = kp[d >> 1];
                h2  p1 = kp[(32 + d) >> 1];
                b0[2 * v] = p0[0]; b0[2 * v + 1] = p0[1];
                b1[2 * v] = p1[0]; b1[2 * v + 1] = p1[1];
            }
            c1 = wmma_f16(aQ0, b0, c1);
            c1 = wmma_f16(aQ1, b1, c1);
        }

        float padk0 = (float)pad[kb + lo];
        float padk1 = (float)pad[kb + 16 + lo];

        // ---- online softmax: m/Z over ALL keys, masks applied to numerator ----
        #pragma unroll
        for (int r = 0; r < 8; ++r) {
            float s0 = c0[r] * 0.125f;               // 1/sqrt(64)
            float s1 = c1[r] * 0.125f;
            float bm = fmaxf(s0, s1);
            #pragma unroll
            for (int off = 1; off < 16; off <<= 1)
                bm = fmaxf(bm, __shfl_xor(bm, off)); // stays inside 16-lane half
            float mnew = fmaxf(mrun[r], bm);
            float fac  = __expf(mrun[r] - mnew);
            mrun[r] = mnew;
            float p0 = __expf(s0 - mnew);
            float p1 = __expf(s1 - mnew);
            float rs = p0 + p1;
            #pragma unroll
            for (int off = 1; off < 16; off <<= 1)
                rs += __shfl_xor(rs, off);
            Zrun[r] = Zrun[r] * fac + rs;            // denominator: unmasked
            #pragma unroll
            for (int j = 0; j < 4; ++j) acc[j][r] *= fac;

            int q = q0 + r + hi * 8;                 // C layout row
            p0 *= (kb + lo      <= q) ? padk0 : 0.0f;  // causal * pad(k)
            p1 *= (kb + 16 + lo <= q) ? padk1 : 0.0f;
            int prow = r + hi * 8;
            P[prow * 32 + lo]      = (_Float16)p0;   // C-layout -> LDS
            P[prow * 32 + lo + 16] = (_Float16)p1;
        }
        asm volatile("s_wait_dscnt 0" ::: "memory"); // P stores visible to reloads

        // ---- reload P in A-fragment layout (in-lane transpose via LDS) ----
        v16h aP;
        #pragma unroll
        for (int v = 0; v < 8; ++v) {
            int kl = kpatA(v, hi);
            h2  p  = *(const h2*)&P[lo * 32 + kl];
            aP[2 * v] = p[0]; aP[2 * v + 1] = p[1];
        }

        // ---- acc += P[16x32] @ V[32x64] : 4 WMMAs over feature tiles ----
        #pragma unroll
        for (int j = 0; j < 4; ++j) {
            const h2* vp = (const h2*)(V + (size_t)(16 * j + lo) * SEQ + kb);
            v16h bV;
            #pragma unroll
            for (int v = 0; v < 8; ++v) {
                int kl = kpatA(v, hi);
                h2  p  = vp[kl >> 1];
                bV[2 * v] = p[0]; bV[2 * v + 1] = p[1];
            }
            acc[j] = wmma_f16(aP, bV, acc[j]);
        }
    }

    // ---- epilogue: out = pad(q) * acc / Z ----
    #pragma unroll
    for (int r = 0; r < 8; ++r) {
        int   q  = q0 + r + hi * 8;
        float sc = (float)pad[q] / Zrun[r];
        #pragma unroll
        for (int j = 0; j < 4; ++j)
            out[((size_t)b * SEQ + q) * D_K + 16 * j + lo] = acc[j][r] * sc;
    }
}

extern "C" void kernel_launch(void* const* d_in, const int* in_sizes, int n_in,
                              void* d_out, int out_size, void* d_ws, size_t ws_size,
                              hipStream_t stream)
{
    const float* x  = (const float*)d_in[0];
    const int*   pm = (const int*)d_in[1];
    const float* Wq = (const float*)d_in[2];
    const float* bq = (const float*)d_in[3];
    const float* Wk = (const float*)d_in[4];
    const float* bk = (const float*)d_in[5];
    const float* Wv = (const float*)d_in[6];
    const float* bv = (const float*)d_in[7];
    float*       out = (float*)d_out;

    // Workspace: Qh (2MB) | Kh (2MB) | Vt transposed (2MB), all f16.
    _Float16* Qh = (_Float16*)d_ws;
    _Float16* Kh = Qh + (size_t)NROWS * D_K;
    _Float16* Vt = Kh + (size_t)NROWS * D_K;

    qkv_proj_kernel<<<NROWS / 64, 128, 0, stream>>>(x, Wq, bq, Wk, bk, Wv, bv,
                                                    Qh, Kh, Vt);
    attn_kernel<<<BATCH * (SEQ / 64), 128, 0, stream>>>(Qh, Kh, Vt, pm, out);
}
